// AttHGT_24661702214217
// MI455X (gfx1250) — matrix operand order repaired and checked by reading, not compiled, not run on hardware
//
#include <hip/hip_runtime.h>
#include <hip/hip_bf16.h>

typedef __attribute__((ext_vector_type(16))) _Float16 v16h;
typedef __attribute__((ext_vector_type(8)))  float    v8f;

static constexpr int cNu = 40000, cNd = 20000;
static constexpr int cE  = 80000, cER = 80000;

// ---------------- device helpers ----------------
__device__ __forceinline__ float geluf(float x) {
    return 0.5f * x * (1.0f + erff(x * 0.70710678118654752f));
}
// order-preserving float<->uint encoding so atomicMax(u32) == float max
__device__ __forceinline__ unsigned fenc(float x) {
    unsigned u = __float_as_uint(x);
    return (u & 0x80000000u) ? ~u : (u | 0x80000000u);
}
__device__ __forceinline__ float fdec(unsigned u) {
    return (u & 0x80000000u) ? __uint_as_float(u & 0x7fffffffu) : __uint_as_float(~u);
}

// ---------------- weight convert: W (B,K,N) f32 -> WT (B,N,K) f16 ----------------
__global__ void k_w2hT(const float* __restrict__ W, _Float16* __restrict__ WT,
                       int B, int K, int N) {
    long i = (long)blockIdx.x * blockDim.x + threadIdx.x;
    long total = (long)B * K * N;
    if (i >= total) return;
    int kn = K * N;
    int b = (int)(i / kn);
    int r = (int)(i % kn);
    int k = r / N, n = r % N;
    WT[(long)b * kn + (long)n * K + k] = (_Float16)W[i];
}

// ---------------- generic batched WMMA GEMM, 16x64 output per wave ----------------
// C[b] = actO( actA(A[b]) @ W[b] + bias ), W given transposed f16 (N x K).
// One wave32 owns a 16(M) x 64(N) strip. Two-stage software pipeline over K
// (requires K % 64 == 0, true for all GEMMs here): stage n+1's A (raw f32) and
// B (f16) are prefetched before stage n's 4 WMMAs issue, and the f32->f16 A
// conversion happens at consume time so the cvt VALU co-executes with the
// XDL WMMA pipe. N % 64 == 0 required (true here).
template <int ACT_A, int ACT_OUT>
__global__ __launch_bounds__(256) void k_gemm(
    const float* __restrict__ A, long lda, long aBS,
    const _Float16* __restrict__ WT, long wBS,
    const float* __restrict__ bias,
    float* __restrict__ C, long ldc, long cBS,
    int M, int N, int K, int totalWaves) {
    int wave = (int)(((long)blockIdx.x * blockDim.x + threadIdx.x) >> 5);
    if (wave >= totalWaves) return;      // wave-uniform, EXEC stays all-ones for WMMA
    int lane = threadIdx.x & 31;
    int tilesN4 = N >> 6, tilesM = M >> 4;
    int tpb = tilesM * tilesN4;
    int b = wave / tpb;
    int t = wave - b * tpb;
    int tm = t / tilesN4, tn4 = t - tm * tilesN4;

    const float*    Ab = A  + (long)b * aBS;
    const _Float16* Wb = WT + (long)b * wBS;
    float*          Cb = C  + (long)b * cBS;

    int r15 = lane & 15, hi = lane >> 4;
    int row = (tm << 4) + r15;          // A row this lane feeds
    int col = (tn4 << 6) + r15;         // first of 4 B cols this lane feeds
    int klo = hi << 4;                  // A: 16 contiguous K at klo
    int wko = hi << 4;                  // B: 16 contiguous K halves at wko

    const float*    arow = Ab + (long)row * lda + klo;
    const _Float16* wcol = Wb + (long)col * K + wko;
    const long K16 = (long)16 * K;

    auto loadA = [&](float4* r, int k) {
        const float4* p = (const float4*)(arow + k);
        r[0] = p[0]; r[1] = p[1];      // K klo..klo+7
        r[2] = p[2]; r[3] = p[3];      // K klo+8..klo+15
    };
    auto loadB = [&](v16h* bb, int k) {
        bb[0] = *(const v16h*)(wcol + k);
        bb[1] = *(const v16h*)(wcol + K16 + k);
        bb[2] = *(const v16h*)(wcol + 2 * K16 + k);
        bb[3] = *(const v16h*)(wcol + 3 * K16 + k);
    };
    auto cvtA = [&](const float4* r) -> v16h {
        float tt[16] = {r[0].x, r[0].y, r[0].z, r[0].w, r[1].x, r[1].y, r[1].z, r[1].w,
                        r[2].x, r[2].y, r[2].z, r[2].w, r[3].x, r[3].y, r[3].z, r[3].w};
        v16h af;
#pragma unroll
        for (int i = 0; i < 16; i++) {
            float x = tt[i];
            if (ACT_A == 1) x = geluf(x);
            af[i] = (_Float16)x;
        }
        return af;
    };

    v8f acc0 = {}, acc1 = {}, acc2 = {}, acc3 = {};
    float4 a0[4], a1[4];
    v16h  b0[4], b1[4];

    loadA(a0, 0);
    loadB(b0, 0);
    for (int k = 0; k < K; k += 64) {
        // prefetch stage 1 (k+32 always < K since K % 64 == 0)
        loadA(a1, k + 32);
        loadB(b1, k + 32);
        {   // consume stage 0: cvt co-executes with WMMA pipe
            v16h af = cvtA(a0);
            acc0 = __builtin_amdgcn_wmma_f32_16x16x32_f16(false, af, false, b0[0], (short)0, acc0, false, false);
            acc1 = __builtin_amdgcn_wmma_f32_16x16x32_f16(false, af, false, b0[1], (short)0, acc1, false, false);
            acc2 = __builtin_amdgcn_wmma_f32_16x16x32_f16(false, af, false, b0[2], (short)0, acc2, false, false);
            acc3 = __builtin_amdgcn_wmma_f32_16x16x32_f16(false, af, false, b0[3], (short)0, acc3, false, false);
        }
        if (k + 64 < K) {  // prefetch stage 0 of next iteration (wave-uniform branch)
            loadA(a0, k + 64);
            loadB(b0, k + 64);
        }
        {   // consume stage 1
            v16h af = cvtA(a1);
            acc0 = __builtin_amdgcn_wmma_f32_16x16x32_f16(false, af, false, b1[0], (short)0, acc0, false, false);
            acc1 = __builtin_amdgcn_wmma_f32_16x16x32_f16(false, af, false, b1[1], (short)0, acc1, false, false);
            acc2 = __builtin_amdgcn_wmma_f32_16x16x32_f16(false, af, false, b1[2], (short)0, acc2, false, false);
            acc3 = __builtin_amdgcn_wmma_f32_16x16x32_f16(false, af, false, b1[3], (short)0, acc3, false, false);
        }
    }

#pragma unroll
    for (int j = 0; j < 4; j++) {
        v8f acc = (j == 0) ? acc0 : (j == 1) ? acc1 : (j == 2) ? acc2 : acc3;
        int cj = col + j * 16;
        float bc = bias ? bias[cj] : 0.0f;
#pragma unroll
        for (int r = 0; r < 8; r++) {
            float v = acc[r] + bc;
            if (ACT_OUT == 1) v = fmaxf(v, 0.0f);
            else if (ACT_OUT == 2) v = tanhf(v);
            int rr = (tm << 4) + r + (hi << 3);  // C layout: VGPR r -> M=r (+8 for hi lanes)
            Cb[(long)rr * ldc + cj] = v;
        }
    }
}

// ---------------- HAN / GAT kernels ----------------
__global__ void k_gat_logits(const float* __restrict__ h, const float* __restrict__ a_s,
                             const float* __restrict__ a_d, float* __restrict__ als,
                             float* __restrict__ ald, int total) {
    int i = blockIdx.x * blockDim.x + threadIdx.x;
    if (i >= total) return;
    int n = i >> 2, hh = i & 3;
    const float* hp = h + (long)n * 64 + hh * 16;
    float s1 = 0.f, s2 = 0.f;
#pragma unroll
    for (int j = 0; j < 16; j++) {
        float v = hp[j];
        s1 += v * a_s[hh * 16 + j];
        s2 += v * a_d[hh * 16 + j];
    }
    als[i] = s1;
    ald[i] = s2;
}

__global__ void k_gat_e1(const int* __restrict__ ei, int E_, const float* __restrict__ als,
                         const float* __restrict__ ald, float* __restrict__ alw,
                         unsigned* __restrict__ mg) {
    int i = blockIdx.x * blockDim.x + threadIdx.x;
    if (i >= E_ * 4) return;
    int e = i >> 2, hh = i & 3;
    int s = ei[e], d = ei[E_ + e];
    float a = als[s * 4 + hh] + ald[d * 4 + hh];
    a = (a > 0.f) ? a : 0.2f * a;  // leaky_relu 0.2
    alw[i] = a;
    atomicMax(&mg[d * 4 + hh], fenc(a));
}

__global__ void k_gat_e2(const int* __restrict__ ei, int E_, const float* __restrict__ h,
                         const float* __restrict__ alw, const unsigned* __restrict__ mg,
                         float* __restrict__ sg, float* __restrict__ og) {
    int gid = blockIdx.x * blockDim.x + threadIdx.x;
    int e = gid >> 5;
    if (e >= E_) return;
    int lane = gid & 31;
    int s = ei[e], d = ei[E_ + e];
    int hh = lane >> 3;
    float ex = expf(alw[e * 4 + hh] - fdec(mg[d * 4 + hh]));
    if ((lane & 7) == 0) atomicAdd(&sg[d * 4 + hh], ex);
    int idx = lane * 2;
    atomicAdd(&og[(long)d * 64 + idx],     h[(long)s * 64 + idx] * ex);
    atomicAdd(&og[(long)d * 64 + idx + 1], h[(long)s * 64 + idx + 1] * ex);
}

__global__ void k_gat_fin(const float* __restrict__ og, const float* __restrict__ sg,
                          float* __restrict__ outs, int total) {
    int i = blockIdx.x * blockDim.x + threadIdx.x;
    if (i >= total) return;
    int n = i >> 6, hh = (i >> 4) & 3;
    outs[i] = fmaxf(og[i] / (sg[n * 4 + hh] + 1e-16f), 0.0f);
}

// column-mean reduction for semantic attention: 128 blocks, one per (r,c)
__global__ void k_col_reduce(const float* __restrict__ t, float* __restrict__ colsum, int rows) {
    __shared__ float red[256];
    int rc = blockIdx.x;
    const float* tp = t + (long)(rc >> 6) * rows * 64 + (rc & 63);
    float s = 0.f;
    for (int n = threadIdx.x; n < rows; n += blockDim.x) s += tp[(long)n * 64];
    red[threadIdx.x] = s;
    __syncthreads();
    for (int st = 128; st > 0; st >>= 1) {
        if (threadIdx.x < st) red[threadIdx.x] += red[threadIdx.x + st];
        __syncthreads();
    }
    if (threadIdx.x == 0) colsum[rc] = red[0];
}

__global__ void k_sem_score(const float* __restrict__ colsum, const float* __restrict__ q_sem,
                            float* __restrict__ sem, float invN) {
    float sc0 = 0.f, sc1 = 0.f;
    for (int c = 0; c < 64; c++) {
        sc0 += q_sem[c] * colsum[c];
        sc1 += q_sem[c] * colsum[64 + c];
    }
    sc0 *= invN; sc1 *= invN;
    float mx = fmaxf(sc0, sc1);
    float e0 = expf(sc0 - mx), e1 = expf(sc1 - mx);
    sem[0] = e0 / (e0 + e1);
    sem[1] = e1 / (e0 + e1);
}

__global__ void k_sem_combine(const float* __restrict__ o0, const float* __restrict__ o1,
                              const float* __restrict__ sem, float* __restrict__ xref, int total) {
    int i = blockIdx.x * blockDim.x + threadIdx.x;
    if (i >= total) return;
    xref[i] = sem[0] * o0[i] + sem[1] * o1[i];
}

// ---------------- HGT edge kernels (dst is always a user after dead-code elim) ----------------
__global__ void k_hgt_e1(const int* __restrict__ ei, int E_, const float* __restrict__ kqvu,
                         const float* __restrict__ kp, const float* __restrict__ p,
                         float* __restrict__ aw, unsigned* __restrict__ m2) {
    int gid = blockIdx.x * blockDim.x + threadIdx.x;
    int e = gid >> 5;
    if (e >= E_) return;
    int lane = gid & 31;
    int s = ei[e], d = ei[E_ + e];
    const float* qp = kqvu + (long)d * 768 + 256 + lane * 8;  // q section of kqv_user
    const float* kq = kp + (long)s * 256 + lane * 8;
    float ps = 0.f;
#pragma unroll
    for (int t = 0; t < 8; t++) ps += qp[t] * kq[t];
    ps += __shfl_xor(ps, 1);
    ps += __shfl_xor(ps, 2);
    ps += __shfl_xor(ps, 4);   // 8-lane head groups (wave32)
    if ((lane & 7) == 0) {
        int hh = lane >> 3;
        float a = ps * p[hh] * 0.125f;  // 1/sqrt(64)
        aw[e * 4 + hh] = a;
        atomicMax(&m2[d * 4 + hh], fenc(a));
    }
}

__global__ void k_hgt_e2(const int* __restrict__ ei, int E_, const float* __restrict__ aw,
                         const unsigned* __restrict__ m2, float* __restrict__ s2,
                         const float* __restrict__ vp, float* __restrict__ oa) {
    int gid = blockIdx.x * blockDim.x + threadIdx.x;
    int e = gid >> 5;
    if (e >= E_) return;
    int lane = gid & 31;
    int s = ei[e], d = ei[E_ + e];
    int hh = lane >> 3;
    float ex = expf(aw[e * 4 + hh] - fdec(m2[d * 4 + hh]));
    if ((lane & 7) == 0) atomicAdd(&s2[d * 4 + hh], ex);
    const float* vq = vp + (long)s * 256 + lane * 8;
    float* op = oa + (long)d * 256 + lane * 8;
#pragma unroll
    for (int t = 0; t < 8; t++) atomicAdd(&op[t], vq[t] * ex);
}

__global__ void k_hgt_fin(float* __restrict__ oa, const float* __restrict__ s2, int total) {
    int i = blockIdx.x * blockDim.x + threadIdx.x;
    if (i >= total) return;
    int n = i >> 8, hh = (i >> 6) & 3;
    oa[i] /= (s2[n * 4 + hh] + 1e-16f);
}

__global__ void k_skip_cat(const float* __restrict__ ou, const float* __restrict__ xu,
                           const float* __restrict__ xref, const float* __restrict__ skip,
                           float* __restrict__ cat, int total) {
    int i = blockIdx.x * blockDim.x + threadIdx.x;
    if (i >= total) return;
    int n = i / 320, c = i - n * 320;
    float su = 1.f / (1.f + expf(-skip[0]));
    cat[i] = (c < 256)
                 ? (su * ou[(long)n * 256 + c] + (1.f - su) * xu[(long)n * 256 + c])
                 : xref[(long)n * 64 + (c - 256)];
}

// ---------------- host-side GEMM dispatch ----------------
static void launch_gemm(int actA, int actO,
                        const float* A, long lda, long aBS,
                        const _Float16* WT, long wBS, const float* bias,
                        float* C, long ldc, long cBS,
                        int M, int N, int K, int B, hipStream_t st) {
    int tw = B * (M >> 4) * (N >> 6);    // one wave per 16x64 strip
    dim3 g((unsigned)((tw + 7) / 8)), bl(256);
    if (actA == 0 && actO == 0)
        k_gemm<0, 0><<<g, bl, 0, st>>>(A, lda, aBS, WT, wBS, bias, C, ldc, cBS, M, N, K, tw);
    else if (actA == 0 && actO == 1)
        k_gemm<0, 1><<<g, bl, 0, st>>>(A, lda, aBS, WT, wBS, bias, C, ldc, cBS, M, N, K, tw);
    else if (actA == 0 && actO == 2)
        k_gemm<0, 2><<<g, bl, 0, st>>>(A, lda, aBS, WT, wBS, bias, C, ldc, cBS, M, N, K, tw);
    else
        k_gemm<1, 0><<<g, bl, 0, st>>>(A, lda, aBS, WT, wBS, bias, C, ldc, cBS, M, N, K, tw);
}

extern "C" void kernel_launch(void* const* d_in, const int* in_sizes, int n_in,
                              void* d_out, int out_size, void* d_ws, size_t ws_size,
                              hipStream_t stream) {
    (void)in_sizes; (void)n_in; (void)out_size; (void)ws_size;
    const float* x_user    = (const float*)d_in[0];
    const float* x_drug    = (const float*)d_in[1];
    const float* x_ref_in  = (const float*)d_in[2];
    const int*   ei_du     = (const int*)d_in[4];
    const int*   ei_uu     = (const int*)d_in[5];
    const int*   ei_r1     = (const int*)d_in[6];
    const int*   ei_r2     = (const int*)d_in[7];
    const float* W_han     = (const float*)d_in[8];
    const float* b_han     = (const float*)d_in[9];
    const float* a_src_r1  = (const float*)d_in[10];
    const float* a_dst_r1  = (const float*)d_in[11];
    const float* a_src_r2  = (const float*)d_in[12];
    const float* a_dst_r2  = (const float*)d_in[13];
    const float* Wk_sem    = (const float*)d_in[14];
    const float* bk_sem    = (const float*)d_in[15];
    const float* q_sem     = (const float*)d_in[16];
    const float* W_in_user = (const float*)d_in[17];
    const float* b_in_user = (const float*)d_in[18];
    const float* W_in_drug = (const float*)d_in[19];
    const float* b_in_drug = (const float*)d_in[20];
    const float* W_kqv_u   = (const float*)d_in[21];
    const float* b_kqv_u   = (const float*)d_in[22];
    const float* W_kqv_d   = (const float*)d_in[23];
    const float* b_kqv_d   = (const float*)d_in[24];
    // d_in[25..27] (ud edge weights) are dead: dst=drug never reaches the output
    const float* Wk_du     = (const float*)d_in[28];
    const float* Wv_du     = (const float*)d_in[29];
    const float* p_du      = (const float*)d_in[30];
    const float* Wk_uu     = (const float*)d_in[31];
    const float* Wv_uu     = (const float*)d_in[32];
    const float* p_uu      = (const float*)d_in[33];
    const float* W_out_u   = (const float*)d_in[34];
    const float* b_out_u   = (const float*)d_in[35];
    const float* skip_u    = (const float*)d_in[38];
    const float* W_fin     = (const float*)d_in[40];
    const float* b_fin     = (const float*)d_in[41];

    char* ws = (char*)d_ws;
    size_t cur = 0;
    auto alloc = [&](size_t bytes) -> size_t {
        size_t o = cur;
        cur = (cur + bytes + 255) & ~(size_t)255;
        return o;
    };
    auto F  = [&](size_t o) -> float*    { return (float*)(ws + o); };
    auto Hh = [&](size_t o) -> _Float16* { return (_Float16*)(ws + o); };
    auto U  = [&](size_t o) -> unsigned* { return (unsigned*)(ws + o); };

    // persistent: f16 transposed weights + x_ref_out + semantic scratch
    size_t o_WhanT  = alloc(128 * 64 * 2);
    size_t o_WksemT = alloc(64 * 64 * 2);
    size_t o_WinUT  = alloc(128 * 256 * 2);
    size_t o_WinDT  = alloc(128 * 256 * 2);
    size_t o_WkqvUT = alloc(256 * 768 * 2);
    size_t o_WkqvDT = alloc(256 * 768 * 2);
    size_t o_WkduT  = alloc(4 * 64 * 64 * 2);
    size_t o_WvduT  = alloc(4 * 64 * 64 * 2);
    size_t o_WkuuT  = alloc(4 * 64 * 64 * 2);
    size_t o_WvuuT  = alloc(4 * 64 * 64 * 2);
    size_t o_WoutUT = alloc(256 * 256 * 2);
    size_t o_WfinT  = alloc(320 * 256 * 2);
    size_t o_xref   = alloc((size_t)cNu * 64 * 4);
    size_t o_sem    = alloc(2 * 4);
    size_t o_colsum = alloc(128 * 4);
    size_t mark = cur;

    // phase A (HAN) scratch
    size_t o_h    = alloc((size_t)cNu * 64 * 4);
    size_t o_als  = alloc((size_t)cNu * 4 * 4);
    size_t o_ald  = alloc((size_t)cNu * 4 * 4);
    size_t o_mg   = alloc((size_t)cNu * 4 * 4);
    size_t o_sg   = alloc((size_t)cNu * 4 * 4);
    size_t o_og   = alloc((size_t)cNu * 64 * 4);
    size_t o_alw  = alloc((size_t)cER * 4 * 4);
    size_t o_outs = alloc((size_t)2 * cNu * 64 * 4);  // outs0 | outs1 contiguous
    size_t o_tb   = alloc((size_t)2 * cNu * 64 * 4);
    (void)o_tb;

    // phase B (HGT) scratch — reuses the phase-A region (stream-ordered)
    cur = mark;
    size_t o_xu   = alloc((size_t)cNu * 256 * 4);
    size_t o_kqvu = alloc((size_t)cNu * 768 * 4);
    size_t o_xd   = alloc((size_t)cNd * 256 * 4);
    size_t o_kd   = alloc((size_t)cNd * 256 * 4);
    size_t o_vd   = alloc((size_t)cNd * 256 * 4);
    size_t o_kpdu = alloc((size_t)cNd * 256 * 4);
    size_t o_vpdu = alloc((size_t)cNd * 256 * 4);
    size_t o_kpuu = alloc((size_t)cNu * 256 * 4);
    size_t o_vpuu = alloc((size_t)cNu * 256 * 4);
    size_t o_aw   = alloc((size_t)2 * cE * 4 * 4);    // du | uu
    size_t o_m2   = alloc((size_t)cNu * 4 * 4);
    size_t o_s2   = alloc((size_t)cNu * 4 * 4);
    size_t o_oa   = alloc((size_t)cNu * 256 * 4);
    size_t o_ou   = alloc((size_t)cNu * 256 * 4);
    size_t o_cat  = alloc((size_t)cNu * 320 * 4);

    auto t2h = [&](const float* W, size_t off, int B, int K, int N) {
        long tot = (long)B * K * N;
        k_w2hT<<<dim3((unsigned)((tot + 255) / 256)), dim3(256), 0, stream>>>(W, Hh(off), B, K, N);
    };
    // weight conversion (small; done every call for determinism)
    t2h(W_han, o_WhanT, 1, 128, 64);
    t2h(Wk_sem, o_WksemT, 1, 64, 64);
    t2h(W_in_user, o_WinUT, 1, 128, 256);
    t2h(W_in_drug, o_WinDT, 1, 128, 256);
    t2h(W_kqv_u, o_WkqvUT, 1, 256, 768);
    t2h(W_kqv_d, o_WkqvDT, 1, 256, 768);
    t2h(Wk_du, o_WkduT, 4, 64, 64);
    t2h(Wv_du, o_WvduT, 4, 64, 64);
    t2h(Wk_uu, o_WkuuT, 4, 64, 64);
    t2h(Wv_uu, o_WvuuT, 4, 64, 64);
    t2h(W_out_u, o_WoutUT, 1, 256, 256);
    t2h(W_fin, o_WfinT, 1, 320, 256);

    // ---------------- Phase A: HAN ----------------
    // h = x_user_ref @ W_han + b_han
    launch_gemm(0, 0, x_ref_in, 128, 0, Hh(o_WhanT), 0, b_han, F(o_h), 64, 0,
                cNu, 64, 128, 1, stream);

    const int* eis[2] = {ei_r1, ei_r2};
    const float* asrc[2] = {a_src_r1, a_src_r2};
    const float* adst[2] = {a_dst_r1, a_dst_r2};
    for (int r = 0; r < 2; r++) {
        k_gat_logits<<<dim3((cNu * 4 + 255) / 256), dim3(256), 0, stream>>>(
            F(o_h), asrc[r], adst[r], F(o_als), F(o_ald), cNu * 4);
        hipMemsetAsync(ws + o_mg, 0, (size_t)cNu * 4 * 4, stream);
        hipMemsetAsync(ws + o_sg, 0, (size_t)cNu * 4 * 4, stream);
        hipMemsetAsync(ws + o_og, 0, (size_t)cNu * 64 * 4, stream);
        k_gat_e1<<<dim3((cER * 4 + 255) / 256), dim3(256), 0, stream>>>(
            eis[r], cER, F(o_als), F(o_ald), F(o_alw), U(o_mg));
        k_gat_e2<<<dim3((cER + 7) / 8), dim3(256), 0, stream>>>(
            eis[r], cER, F(o_h), F(o_alw), U(o_mg), F(o_sg), F(o_og));
        k_gat_fin<<<dim3((cNu * 64 + 255) / 256), dim3(256), 0, stream>>>(
            F(o_og), F(o_sg), F(o_outs) + (size_t)r * cNu * 64, cNu * 64);
    }
    // semantic attention: tanh(outs @ Wk_sem + bk) -> column-mean -> softmax over 2
    launch_gemm(0, 2, F(o_outs), 64, (long)cNu * 64, Hh(o_WksemT), 0, bk_sem,
                F(o_tb), 64, (long)cNu * 64, cNu, 64, 64, 2, stream);
    k_col_reduce<<<dim3(128), dim3(256), 0, stream>>>(F(o_tb), F(o_colsum), cNu);
    k_sem_score<<<dim3(1), dim3(1), 0, stream>>>(F(o_colsum), q_sem, F(o_sem), 1.0f / cNu);
    k_sem_combine<<<dim3((cNu * 64 + 255) / 256), dim3(256), 0, stream>>>(
        F(o_outs), F(o_outs) + (size_t)cNu * 64, F(o_sem), F(o_xref), cNu * 64);

    // ---------------- Phase B: HGT (user output path only) ----------------
    launch_gemm(0, 1, x_user, 128, 0, Hh(o_WinUT), 0, b_in_user, F(o_xu), 256, 0,
                cNu, 256, 128, 1, stream);
    launch_gemm(0, 1, x_drug, 128, 0, Hh(o_WinDT), 0, b_in_drug, F(o_xd), 256, 0,
                cNd, 256, 128, 1, stream);
    // kqv_user full (k,q,v all needed); drug: only k and v sections
    launch_gemm(0, 0, F(o_xu), 256, 0, Hh(o_WkqvUT), 0, b_kqv_u, F(o_kqvu), 768, 0,
                cNu, 768, 256, 1, stream);
    launch_gemm(0, 0, F(o_xd), 256, 0, Hh(o_WkqvDT), 0, b_kqv_d, F(o_kd), 256, 0,
                cNd, 256, 256, 1, stream);
    launch_gemm(0, 0, F(o_xd), 256, 0, Hh(o_WkqvDT) + (size_t)512 * 256, 0, b_kqv_d + 512,
                F(o_vd), 256, 0, cNd, 256, 256, 1, stream);
    // per-head 64x64 transforms, batched over H=4
    launch_gemm(0, 0, F(o_kd), 256, 64, Hh(o_WkduT), 4096, nullptr, F(o_kpdu), 256, 64,
                cNd, 64, 64, 4, stream);
    launch_gemm(0, 0, F(o_vd), 256, 64, Hh(o_WvduT), 4096, nullptr, F(o_vpdu), 256, 64,
                cNd, 64, 64, 4, stream);
    launch_gemm(0, 0, F(o_kqvu), 768, 64, Hh(o_WkuuT), 4096, nullptr, F(o_kpuu), 256, 64,
                cNu, 64, 64, 4, stream);
    launch_gemm(0, 0, F(o_kqvu) + 512, 768, 64, Hh(o_WvuuT), 4096, nullptr, F(o_vpuu), 256, 64,
                cNu, 64, 64, 4, stream);

    hipMemsetAsync(ws + o_m2, 0, (size_t)cNu * 4 * 4, stream);
    hipMemsetAsync(ws + o_s2, 0, (size_t)cNu * 4 * 4, stream);
    hipMemsetAsync(ws + o_oa, 0, (size_t)cNu * 256 * 4, stream);

    float* aw_du = F(o_aw);
    float* aw_uu = F(o_aw) + (size_t)cE * 4;
    dim3 eg((cE + 7) / 8), eb(256);
    k_hgt_e1<<<eg, eb, 0, stream>>>(ei_du, cE, F(o_kqvu), F(o_kpdu), p_du, aw_du, U(o_m2));
    k_hgt_e1<<<eg, eb, 0, stream>>>(ei_uu, cE, F(o_kqvu), F(o_kpuu), p_uu, aw_uu, U(o_m2));
    k_hgt_e2<<<eg, eb, 0, stream>>>(ei_du, cE, aw_du, U(o_m2), F(o_s2), F(o_vpdu), F(o_oa));
    k_hgt_e2<<<eg, eb, 0, stream>>>(ei_uu, cE, aw_uu, U(o_m2), F(o_s2), F(o_vpuu), F(o_oa));
    k_hgt_fin<<<dim3((cNu * 256 + 255) / 256), dim3(256), 0, stream>>>(
        F(o_oa), F(o_s2), cNu * 256);

    // ou = gelu(out_user) @ W_out_user + b (gelu fused into A-fragment load)
    launch_gemm(1, 0, F(o_oa), 256, 0, Hh(o_WoutUT), 0, b_out_u, F(o_ou), 256, 0,
                cNu, 256, 256, 1, stream);
    // gated skip + concat with x_ref_out
    k_skip_cat<<<dim3((cNu * 320 + 255) / 256), dim3(256), 0, stream>>>(
        F(o_ou), F(o_xu), F(o_xref), skip_u, F(o_cat), cNu * 320);
    // final: x_emb = cat @ W_fin + b_fin  -> d_out
    launch_gemm(0, 0, F(o_cat), 320, 0, Hh(o_WfinT), 0, b_fin, (float*)d_out, 256, 0,
                cNu, 256, 320, 1, stream);
}
